// SceneGCN_87351044866369
// MI455X (gfx1250) — matrix operand order repaired and checked by reading, not compiled, not run on hardware
//
#include <hip/hip_runtime.h>
#include <hip/hip_bf16.h>

// ---------------------------------------------------------------------------
// SceneGCN on MI455X (gfx1250, wave32, WMMA).
// ~730 GFLOP of GEMMs over 2 steps vs ~1.3 GB HBM traffic (~55us @ 23.3TB/s)
// -> matrix-core bound. All five GEMM shapes run through one templated
// 128x128x32 bf16-WMMA kernel (f32 accumulate), software-pipelined:
//   store_lds(kt) -> barrier -> global_load(kt+1) [no wait] -> 8x WMMA(kt)
// so global latency hides behind v_wmma_f32_16x16x32_bf16 bursts.
// fp32->bf16 pair conversion is a vector fptrunc so the backend emits a
// single v_cvt_pk_bf16_f32 per 2 elements (no scalar repack chain).
// Denominators are precomputed as reciprocals so epilogues are pure FMA.
// ---------------------------------------------------------------------------

typedef __attribute__((ext_vector_type(16))) __bf16 v16bf;
typedef __attribute__((ext_vector_type(8)))  float  v8f;
typedef __attribute__((ext_vector_type(2)))  float  vf2;
typedef __attribute__((ext_vector_type(2)))  __bf16 vbf2;

static constexpr int N_OBJ = 2048;
static constexpr int N_REL = 16384;
static constexpr int DIM   = 1024;
static constexpr float EPSV = 1e-7f;

// native fp32 -> bf16 (hardware convert on gfx1250)
__device__ __forceinline__ unsigned short f2bf(float f) {
    union { __bf16 h; unsigned short u; } cv;
    cv.h = (__bf16)f;
    return cv.u;
}

// packed pair convert: one v_cvt_pk_bf16_f32 per call
__device__ __forceinline__ unsigned pkbf(float x, float y) {
    vf2 f; f.x = x; f.y = y;
    vbf2 h = __builtin_convertvector(f, vbf2);
    union { vbf2 h2; unsigned u; } cv;
    cv.h2 = h;
    return cv.u;
}

__device__ __forceinline__ uint4 pack2f4(float4 a, float4 b) {
    uint4 r;
    r.x = pkbf(a.x, a.y);
    r.y = pkbf(a.z, a.w);
    r.z = pkbf(b.x, b.y);
    r.w = pkbf(b.z, b.w);
    return r;
}

union Frag16 { v16bf v; uint4 q[2]; };

struct Stage { float4 a[4]; float4 b[4]; };

// ---------------------------------------------------------------------------
// C[M,N] = op(A[M,K] x B[K,N]) with bf16 WMMA, f32 accumulation.
//   TRANS_A=1 : logical A[m][k] = Aphys[k][m]   (lda = physical row stride)
//   TRANS_B=1 : logical B[k][n] = Bphys[n][k]   (ldb = physical row stride)
//   EPI=0     : C = relu(acc + aux[n])          (aux = bias)
//   EPI=1     : C += coef * acc * aux[m]        (aux = 1/(rowsum+eps))
// Block tile 128x128, K-tile 32. 256 threads = 8 waves in a 2(M)x4(N) grid,
// each wave computes a 64x32 patch = 4x2 WMMA accumulators (64 VGPRs).
// Double-buffered LDS + register staging: one barrier per K-step.
// ---------------------------------------------------------------------------
template <int TRANS_A, int TRANS_B, int EPI>
__global__ __launch_bounds__(256)
void wmma_gemm_kernel(const float* __restrict__ A, int lda,
                      const float* __restrict__ B, int ldb,
                      float* __restrict__ C, int ldc,
                      int K, const float* __restrict__ aux, float coef) {
    __shared__ unsigned short Ash[2][128][32];   // [buf][m][k] bf16
    __shared__ unsigned short Bsh[2][128][32];   // [buf][n][k] bf16 (n-major)

    const int tid  = threadIdx.x;
    const int lane = tid & 31;
    const int wave = tid >> 5;
    const int wm   = wave & 1;    // 2 waves along M
    const int wn   = wave >> 1;   // 4 waves along N
    const int m0   = blockIdx.y * 128;
    const int n0   = blockIdx.x * 128;

    // ---- global tile fetch into registers (issue; wait happens at store) ----
    auto load_global = [&](int kt, Stage& st) {
        const int k0 = kt * 32;
        if constexpr (!TRANS_A) {
            const int row = tid >> 1, half = tid & 1;
            const float4* s4 =
                (const float4*)(A + (size_t)(m0 + row) * lda + k0 + half * 16);
            st.a[0] = s4[0]; st.a[1] = s4[1]; st.a[2] = s4[2]; st.a[3] = s4[3];
            __builtin_prefetch((const float*)s4 + 32, 0, 0);
        } else {
            const int k = tid >> 3, seg = tid & 7;
            const float* src = A + (size_t)(k0 + k) * lda + m0 + seg * 16;
            const float4* s4 = (const float4*)src;
            st.a[0] = s4[0]; st.a[1] = s4[1]; st.a[2] = s4[2]; st.a[3] = s4[3];
            __builtin_prefetch(src + (size_t)32 * lda, 0, 0);
        }
        if constexpr (TRANS_B) {   // B physical = W[N,K] row-major
            const int n = tid >> 1, half = tid & 1;
            const float4* s4 =
                (const float4*)(B + (size_t)(n0 + n) * ldb + k0 + half * 16);
            st.b[0] = s4[0]; st.b[1] = s4[1]; st.b[2] = s4[2]; st.b[3] = s4[3];
            __builtin_prefetch((const float*)s4 + 32, 0, 0);
        } else {                   // B physical = F[K,N] row-major
            const int k = tid >> 3, seg = tid & 7;
            const float* src = B + (size_t)(k0 + k) * ldb + n0 + seg * 16;
            const float4* s4 = (const float4*)src;
            st.b[0] = s4[0]; st.b[1] = s4[1]; st.b[2] = s4[2]; st.b[3] = s4[3];
            __builtin_prefetch(src + (size_t)32 * ldb, 0, 0);
        }
    };

    // ---- convert + store staged tile into LDS buffer ----
    auto store_lds = [&](int buf, const Stage& st) {
        if constexpr (!TRANS_A) {
            const int row = tid >> 1, half = tid & 1;
            uint4* dst = (uint4*)&Ash[buf][row][half * 16];
            dst[0] = pack2f4(st.a[0], st.a[1]);
            dst[1] = pack2f4(st.a[2], st.a[3]);
        } else {
            const int k = tid >> 3, seg = tid & 7;
#pragma unroll
            for (int q = 0; q < 4; ++q) {
                Ash[buf][seg * 16 + q * 4 + 0][k] = f2bf(st.a[q].x);
                Ash[buf][seg * 16 + q * 4 + 1][k] = f2bf(st.a[q].y);
                Ash[buf][seg * 16 + q * 4 + 2][k] = f2bf(st.a[q].z);
                Ash[buf][seg * 16 + q * 4 + 3][k] = f2bf(st.a[q].w);
            }
        }
        if constexpr (TRANS_B) {
            const int n = tid >> 1, half = tid & 1;
            uint4* dst = (uint4*)&Bsh[buf][n][half * 16];
            dst[0] = pack2f4(st.b[0], st.b[1]);
            dst[1] = pack2f4(st.b[2], st.b[3]);
        } else {
            const int k = tid >> 3, seg = tid & 7;
#pragma unroll
            for (int q = 0; q < 4; ++q) {
                Bsh[buf][seg * 16 + q * 4 + 0][k] = f2bf(st.b[q].x);
                Bsh[buf][seg * 16 + q * 4 + 1][k] = f2bf(st.b[q].y);
                Bsh[buf][seg * 16 + q * 4 + 2][k] = f2bf(st.b[q].z);
                Bsh[buf][seg * 16 + q * 4 + 3][k] = f2bf(st.b[q].w);
            }
        }
    };

    v8f acc[4][2];
#pragma unroll
    for (int i = 0; i < 4; ++i)
#pragma unroll
        for (int j = 0; j < 2; ++j)
#pragma unroll
            for (int e = 0; e < 8; ++e) acc[i][j][e] = 0.0f;

    // fragment index precompute (ISA 16-bit operand layouts):
    // A 16x32: lane<16 -> K 0..7 & 16..23 ; lane>=16 -> K 8..15 & 24..31
    // B 32x16: lane<16 -> K 0..7 & 16..23 ; lane>=16 -> K 8..15 & 24..31
    const int m_in = lane & 15;
    const int ksel = lane >> 4;
    const int a_kb = ksel * 8;
    const int b_kb = ksel * 8;

    const int nk = K >> 5;
    Stage st;
    load_global(0, st);

    for (int kt = 0; kt < nk; ++kt) {
        const int cur = kt & 1;
        store_lds(cur, st);          // waits on this tile's global loads
        __syncthreads();             // buffer `cur` fully populated
        if (kt + 1 < nk) load_global(kt + 1, st);   // issue next tile, no wait

        Frag16 fa[4], fb[2];
#pragma unroll
        for (int i = 0; i < 4; ++i) {
            const int r = wm * 64 + i * 16 + m_in;
            fa[i].q[0] = *(const uint4*)&Ash[cur][r][a_kb];
            fa[i].q[1] = *(const uint4*)&Ash[cur][r][a_kb + 16];
        }
#pragma unroll
        for (int j = 0; j < 2; ++j) {
            const int r = wn * 32 + j * 16 + m_in;
            fb[j].q[0] = *(const uint4*)&Bsh[cur][r][b_kb];
            fb[j].q[1] = *(const uint4*)&Bsh[cur][r][b_kb + 16];
        }

#pragma unroll
        for (int i = 0; i < 4; ++i)
#pragma unroll
            for (int j = 0; j < 2; ++j)
                acc[i][j] = __builtin_amdgcn_wmma_f32_16x16x32_bf16(
                    false, fa[i].v, false, fb[j].v,
                    (short)0, acc[i][j], false, false);
    }

    // ---- epilogue (C/D layout: VGPR r -> M = r + (lane<16 ? 0 : 8)) ----
    const int n_in = lane & 15;
    const int mb   = (lane >> 4) << 3;
#pragma unroll
    for (int i = 0; i < 4; ++i) {
#pragma unroll
        for (int j = 0; j < 2; ++j) {
            const int gn = n0 + wn * 32 + j * 16 + n_in;
#pragma unroll
            for (int r = 0; r < 8; ++r) {
                const int gm = m0 + wm * 64 + i * 16 + mb + r;
                float a = acc[i][j][r];
                float* p = C + (size_t)gm * ldc + gn;
                if constexpr (EPI == 0) {
                    float v = a + aux[gn];
                    *p = v > 0.0f ? v : 0.0f;
                } else {
                    *p = fmaf(coef * a, aux[gm], *p);   // aux = 1/denom
                }
            }
        }
    }
}

// out[row] = 1 / (sum_n A[row][n] + eps)   (one block per row)
__global__ __launch_bounds__(256)
void rowsum_inv_kernel(const float* __restrict__ A, int ncols,
                       float* __restrict__ out) {
    const int row = blockIdx.x;
    const float4* p = (const float4*)(A + (size_t)row * ncols);
    const int n4 = ncols >> 2;
    float s = 0.0f;
    for (int i = threadIdx.x; i < n4; i += 256) {
        float4 v = p[i];
        s += v.x + v.y + v.z + v.w;
    }
    __shared__ float red[256];
    red[threadIdx.x] = s;
    __syncthreads();
    for (int off = 128; off > 0; off >>= 1) {
        if (threadIdx.x < off) red[threadIdx.x] += red[threadIdx.x + off];
        __syncthreads();
    }
    if (threadIdx.x == 0) out[row] = 1.0f / (red[0] + EPSV);
}

// out[c] = 1 / (sum_r A[r][c] + eps)  (coalesced: thread-per-column)
__global__ __launch_bounds__(256)
void colsum_inv_kernel(const float* __restrict__ A, int nrows, int ncols,
                       float* __restrict__ out) {
    const int c = blockIdx.x * 256 + threadIdx.x;
    if (c >= ncols) return;
    float s = 0.0f;
    for (int r = 0; r < nrows; ++r) s += A[(size_t)r * ncols + c];
    out[c] = 1.0f / (s + EPSV);
}

extern "C" void kernel_launch(void* const* d_in, const int* in_sizes, int n_in,
                              void* d_out, int out_size, void* d_ws, size_t ws_size,
                              hipStream_t stream) {
    (void)in_sizes; (void)n_in; (void)out_size; (void)ws_size;

    const float* obj_feats = (const float*)d_in[0];
    const float* rel_feats = (const float*)d_in[1];
    const float* attn_sub  = (const float*)d_in[2];
    const float* attn_obj  = (const float*)d_in[3];
    const float* attn_self = (const float*)d_in[4];
    const float* W_r2o_sub = (const float*)d_in[5];
    const float* b_r2o_sub = (const float*)d_in[6];
    const float* W_r2o_obj = (const float*)d_in[7];
    const float* b_r2o_obj = (const float*)d_in[8];
    const float* W_o2r_sub = (const float*)d_in[9];
    const float* b_o2r_sub = (const float*)d_in[10];
    const float* W_o2r_obj = (const float*)d_in[11];
    const float* b_o2r_obj = (const float*)d_in[12];
    const float* W_o2o     = (const float*)d_in[13];
    const float* b_o2o     = (const float*)d_in[14];

    // Output buffers double as the in-place state: x_obj then x_rel, fp32.
    float* x_obj = (float*)d_out;
    float* x_rel = x_obj + (size_t)N_OBJ * DIM;

    // Workspace carve-out (~92 MB)
    float* ws      = (float*)d_ws;
    float* fc_big  = ws;                                   // [N_REL, DIM]
    float* fc_self = fc_big  + (size_t)N_REL * DIM;        // [N_OBJ, DIM]
    float* fc_rs   = fc_self + (size_t)N_OBJ * DIM;        // [N_OBJ, DIM]
    float* fc_ro   = fc_rs   + (size_t)N_OBJ * DIM;        // [N_OBJ, DIM]
    float* i_sub   = fc_ro   + (size_t)N_OBJ * DIM;        // [N_OBJ]
    float* i_obj   = i_sub   + N_OBJ;                      // [N_OBJ]
    float* i_self  = i_obj   + N_OBJ;                      // [N_OBJ]
    float* i_subT  = i_self  + N_OBJ;                      // [N_REL]
    float* i_objT  = i_subT  + N_REL;                      // [N_REL]

    // init state from inputs (d2d, capture-safe)
    hipMemcpyAsync(x_obj, obj_feats, (size_t)N_OBJ * DIM * sizeof(float),
                   hipMemcpyDeviceToDevice, stream);
    hipMemcpyAsync(x_rel, rel_feats, (size_t)N_REL * DIM * sizeof(float),
                   hipMemcpyDeviceToDevice, stream);

    // attention normalizers are step-invariant: compute reciprocals once
    rowsum_inv_kernel<<<N_OBJ, 256, 0, stream>>>(attn_sub,  N_REL, i_sub);
    rowsum_inv_kernel<<<N_OBJ, 256, 0, stream>>>(attn_obj,  N_REL, i_obj);
    rowsum_inv_kernel<<<N_OBJ, 256, 0, stream>>>(attn_self, N_OBJ, i_self);
    colsum_inv_kernel<<<N_REL / 256, 256, 0, stream>>>(attn_sub, N_OBJ, N_REL, i_subT);
    colsum_inv_kernel<<<N_REL / 256, 256, 0, stream>>>(attn_obj, N_OBJ, N_REL, i_objT);

    const dim3 blk(256);

    auto linear = [&](const float* X, int M, const float* W, const float* bias,
                      float* out) {
        dim3 g(DIM / 128, M / 128);
        wmma_gemm_kernel<0, 1, 0><<<g, blk, 0, stream>>>(
            X, DIM, W, DIM, out, DIM, DIM, bias, 0.0f);
    };
    auto collect = [&](const float* Attn, int M, int K, const float* F,
                       const float* invd, float coef, float* out) {
        dim3 g(DIM / 128, M / 128);
        wmma_gemm_kernel<0, 0, 1><<<g, blk, 0, stream>>>(
            Attn, K, F, DIM, out, DIM, K, invd, coef);
    };
    auto collectT = [&](const float* Attn, int M, int K, const float* F,
                        const float* invd, float coef, float* out) {
        // logical A[m][k] = Attn[k][m]; physical row stride of Attn is M
        dim3 g(DIM / 128, M / 128);
        wmma_gemm_kernel<1, 0, 1><<<g, blk, 0, stream>>>(
            Attn, M, F, DIM, out, DIM, K, invd, coef);
    };

    const float third = 1.0f / 3.0f;
    const float half  = 0.5f;

    for (int step = 0; step < 2; ++step) {
        // fc's that depend on OLD x_obj — before x_obj is touched
        linear(x_obj, N_OBJ, W_o2o,     b_o2o,     fc_self);
        linear(x_obj, N_OBJ, W_o2r_sub, b_o2r_sub, fc_rs);
        linear(x_obj, N_OBJ, W_o2r_obj, b_o2r_obj, fc_ro);

        // fc's from OLD x_rel feed the object update (x_rel untouched here)
        linear(x_rel, N_REL, W_r2o_sub, b_r2o_sub, fc_big);
        collect(attn_sub, N_OBJ, N_REL, fc_big, i_sub, third, x_obj);
        linear(x_rel, N_REL, W_r2o_obj, b_r2o_obj, fc_big);
        collect(attn_obj, N_OBJ, N_REL, fc_big, i_obj, third, x_obj);
        collect(attn_self, N_OBJ, N_OBJ, fc_self, i_self, third, x_obj);

        // relation update from OLD x_obj (already baked into fc_rs/fc_ro)
        collectT(attn_sub, N_REL, N_OBJ, fc_rs, i_subT, half, x_rel);
        collectT(attn_obj, N_REL, N_OBJ, fc_ro, i_objT, half, x_rel);
    }
}
